// GFN_13408887899007
// MI455X (gfx1250) — compile-verified, matrix operand order checked by reading, not run on hardware
//
#include <hip/hip_runtime.h>
#include <hip/hip_bf16.h>
#include <math.h>

// ---------------- problem constants ----------------
constexpr int DIMC  = 64;
constexpr int BC    = 1024;
constexpr int TC    = 100;
constexpr int HIDC  = 512;
constexpr int TDIMC = 128;
constexpr int HARMC = 64;

constexpr float DT_F      = 1.0f / 100.0f;
constexpr float SQRT_DT_F = 0.1f;
constexpr float PI_F      = 3.14159265358979323846f;
constexpr float TWO_PI_F  = 6.28318530717958647692f;
constexpr float SQRT_2PI_F = 2.5066282746310002f;
constexpr float LOG_VAR_RANGE_F = 4.0f;
constexpr int   N_WRAP_C  = 10;

// batch tile per block
constexpr int BT = 32;           // rows per block
constexpr int NBLK = BC / BT;    // 32 blocks

// ---------------- vector types ----------------
typedef __attribute__((ext_vector_type(8)))  float   v8f;
typedef __attribute__((ext_vector_type(8)))  __bf16  v8bf;
typedef __attribute__((ext_vector_type(16))) __bf16  v16bf;

__device__ inline v16bf cat8(v8bf lo, v8bf hi) {
  return __builtin_shufflevector(lo, hi, 0,1,2,3,4,5,6,7,8,9,10,11,12,13,14,15);
}

// ---------------- math helpers ----------------
__device__ inline float fast_tanh(float x) {
  float e = __expf(2.0f * x);
  return (e - 1.0f) / (e + 1.0f);
}
__device__ inline float gelu_f(float x) {
  float x3 = x * x * x;
  return 0.5f * x * (1.0f + fast_tanh(0.7978845608028654f * (x + 0.044715f * x3)));
}
__device__ inline float wrap_f(float x) {
  float y = x + PI_F;
  y -= TWO_PI_F * floorf(y * (1.0f / TWO_PI_F));
  return y - PI_F;
}
// log prob of wrapped normal: logsumexp over 2*N_WRAP+1 components (one dim)
__device__ inline float wrapped_lp(float delta, float mean, float stdv) {
  float inv = 1.0f / stdv;
  float m = -1e30f, ssum = 0.0f;
  #pragma unroll
  for (int k = -N_WRAP_C; k <= N_WRAP_C; ++k) {
    float z = (delta + (float)k * TWO_PI_F - mean) * inv;
    float c = -0.5f * z * z;
    if (c > m) { ssum = ssum * __expf(m - c) + 1.0f; m = c; }
    else       { ssum += __expf(c - m); }
  }
  return m + __logf(ssum) - __logf(stdv * SQRT_2PI_F);
}

// ---------------- workspace layout (bf16 elements) ----------------
constexpr size_t WS_WST   = 0;                         // [512][64]
constexpr size_t WS_WJ1T  = WS_WST  + 512 * 64;        // [512][640]
constexpr size_t WS_WJ2T  = WS_WJ1T + 512 * 640;       // [512][512]
constexpr size_t WS_WJ3T  = WS_WJ2T + 512 * 512;       // [128][512]
constexpr size_t WS_TE    = WS_WJ3T + 128 * 512;       // [100][128]
constexpr size_t WS_END   = WS_TE + (size_t)TC * TDIMC;
constexpr size_t WS_BYTES = WS_END * 2;

// =====================================================================
// Prep kernel 1: transpose+convert fp32 weight [K][N] -> bf16 [N][K]
// =====================================================================
__global__ void k_transpose_bf16(const float* __restrict__ src,
                                 __bf16* __restrict__ dst, int K, int N) {
  int total = K * N;
  for (int idx = blockIdx.x * blockDim.x + threadIdx.x; idx < total;
       idx += gridDim.x * blockDim.x) {
    int k = idx / N, n = idx - k * N;
    dst[(size_t)n * K + k] = (__bf16)src[(size_t)k * N + n];
  }
}

// =====================================================================
// Prep kernel 2: time embedding te_all[t][128] (batch independent)
// =====================================================================
__global__ __launch_bounds__(128) void k_time_embed(
    const float* __restrict__ Wt1, const float* __restrict__ bt1,
    const float* __restrict__ Wt2, const float* __restrict__ bt2,
    __bf16* __restrict__ teB) {
  __shared__ float tf[2 * HARMC];
  __shared__ float hid[HIDC];
  int t = blockIdx.x;
  int j = threadIdx.x;
  float tval = (float)t * DT_F;
  if (j < 64) tf[j] = __sinf(TWO_PI_F * (float)(j + 1) * tval);
  else        tf[j] = __cosf(TWO_PI_F * (float)(j - 63) * tval);
  __syncthreads();
  #pragma unroll
  for (int r = 0; r < 4; ++r) {
    int h = j + r * 128;
    float acc = bt1[h];
    for (int k = 0; k < 128; ++k) acc += tf[k] * Wt1[k * HIDC + h];
    hid[h] = gelu_f(acc);
  }
  __syncthreads();
  float acc = bt2[j];
  for (int k = 0; k < HIDC; ++k) acc += hid[k] * Wt2[k * TDIMC + j];
  teB[(size_t)t * TDIMC + j] = (__bf16)acc;
}

// =====================================================================
// WMMA GEMM layer: D[BT x ND] = act(A[BT x KD] @ W[KD x ND] + bias)
//   A: LDS bf16 row-major, stride sA
//   Wt: global bf16, pre-transposed [ND][KD] (K contiguous per column)
//   8 waves, each owns 2 M-tiles x NTW N-tiles
//   K-loop unrolled x2 only: keeps live ranges below the 256-VGPR
//   direct window (avoids s_set_vgpr_msb churn) while still keeping
//   ~16 WMMAs + 2 fragment load groups in flight.
// =====================================================================
template <int KD, int ND, int NTW, bool ACT, bool OUTF32>
__device__ void gemm_layer(const __bf16* __restrict__ A, int sA,
                           const __bf16* __restrict__ Wt,
                           const float* __restrict__ bias,
                           __bf16* __restrict__ D, float* __restrict__ Df,
                           int sD) {
  const int tid  = threadIdx.x;
  const int wave = tid >> 5;
  const int lane = tid & 31;
  const int m    = lane & 15;
  const int kh   = (lane >> 4) & 1;     // K-half select per WMMA layout
  const int nb0  = wave * NTW * 16;

  v8f acc[2][NTW];
  #pragma unroll
  for (int i = 0; i < 2; ++i)
    #pragma unroll
    for (int j = 0; j < NTW; ++j) acc[i][j] = (v8f){0,0,0,0,0,0,0,0};

  #pragma unroll 2
  for (int kb = 0; kb < KD; kb += 32) {
    const int c0 = kb + kh * 8;
    // A fragments (two M-tiles), each: two contiguous 16B LDS reads
    const __bf16* ap0 = A + (size_t)m * sA + c0;
    const __bf16* ap1 = A + (size_t)(16 + m) * sA + c0;
    v16bf a0 = cat8(*(const v8bf*)ap0, *(const v8bf*)(ap0 + 16));
    v16bf a1 = cat8(*(const v8bf*)ap1, *(const v8bf*)(ap1 + 16));
    #pragma unroll
    for (int j = 0; j < NTW; ++j) {
      const int n = nb0 + j * 16 + m;
      const __bf16* wp = Wt + (size_t)n * KD + c0;   // L2-resident weights
      v16bf b = cat8(*(const v8bf*)wp, *(const v8bf*)(wp + 16));
      acc[0][j] = __builtin_amdgcn_wmma_f32_16x16x32_bf16(
          false, a0, false, b, (short)0, acc[0][j], false, false);
      acc[1][j] = __builtin_amdgcn_wmma_f32_16x16x32_bf16(
          false, a1, false, b, (short)0, acc[1][j], false, false);
    }
  }

  // epilogue: bias + (gelu) + store to LDS (bf16 or f32)
  #pragma unroll
  for (int mt = 0; mt < 2; ++mt) {
    #pragma unroll
    for (int j = 0; j < NTW; ++j) {
      const int col = nb0 + j * 16 + m;
      const float bcol = bias[col];
      #pragma unroll
      for (int r = 0; r < 8; ++r) {
        const int row = mt * 16 + kh * 8 + r;
        float v = acc[mt][j][r] + bcol;
        if constexpr (ACT) v = gelu_f(v);
        if constexpr (OUTF32) Df[(size_t)row * sD + col] = v;
        else                  D[(size_t)row * sD + col] = (__bf16)v;
      }
    }
  }
}

// =====================================================================
// Main persistent kernel: sequential scan over T steps
// =====================================================================
__global__ __launch_bounds__(256) void k_scan(
    const float* __restrict__ s_in, const float* __restrict__ eps,
    const __bf16* __restrict__ WsT,  const float* __restrict__ bs,
    const __bf16* __restrict__ Wj1T, const float* __restrict__ bj1,
    const __bf16* __restrict__ Wj2T, const float* __restrict__ bj2,
    const __bf16* __restrict__ Wj3T, const float* __restrict__ bj3,
    const __bf16* __restrict__ teB,  const float* __restrict__ flowp,
    float* __restrict__ states, float* __restrict__ gLogpf,
    float* __restrict__ gLogpb, float* __restrict__ gLogf) {
  extern __shared__ __align__(16) char smem[];
  float*  Sbuf = (float*)smem;                    // [32][64]  f32   8KB
  __bf16* Sbf  = (__bf16*)(smem + 8192);          // [32][64]  bf16  4KB
  __bf16* Hbuf = (__bf16*)(smem + 12288);         // [32][640] bf16 40KB
  __bf16* H1   = (__bf16*)(smem + 53248);         // [32][512] bf16 32KB
  __bf16* H2   = (__bf16*)(smem + 86016);         // [32][512] bf16 32KB
  float*  Pf   = (float*)(smem + 118784);         // [32][128] f32  16KB

  const int tid = threadIdx.x;
  const int r0  = blockIdx.x * BT;
  const float flow = flowp[0];

  // one-time prefetch sweep of the bf16 weight arena into near caches
  // (emits global_prefetch_b8; weights then live in WGP$/L2 for all steps)
  {
    const char* wb = (const char*)WsT;
    for (size_t off = (size_t)tid * 128; off < WS_BYTES; off += 256u * 128u)
      __builtin_prefetch(wb + off, 0, 1);
  }

  // init: load state tile, emit states[:,0,:], zero logf[:,T]
  for (int idx = tid; idx < BT * DIMC; idx += 256) {
    int row = idx >> 6, d = idx & 63;
    float sv = s_in[(size_t)(r0 + row) * DIMC + d];
    Sbuf[idx] = sv;
    states[((size_t)(r0 + row) * (TC + 1)) * DIMC + d] = sv;
  }
  if (tid < BT) gLogf[(size_t)(r0 + tid) * (TC + 1) + TC] = 0.0f;
  __syncthreads();

  const int erow = tid >> 3;            // row handled in elementwise tail
  const int edg  = (tid & 7) * 8;       // first dim handled

  for (int t = 0; t < TC; ++t) {
    // hoisted eps loads: issue now, consumed after 4 GEMMs (latency free)
    float er[8];
    {
      const float* ep = eps + (size_t)t * (BC * DIMC) +
                        (size_t)(r0 + erow) * DIMC + edg;
      float4 e0 = *(const float4*)ep;
      float4 e1 = *(const float4*)(ep + 4);
      er[0] = e0.x; er[1] = e0.y; er[2] = e0.z; er[3] = e0.w;
      er[4] = e1.x; er[5] = e1.y; er[6] = e1.z; er[7] = e1.w;
    }

    // stage 0: s -> bf16 A1; te[t] -> Hbuf columns [512..639]
    for (int idx = tid; idx < BT * DIMC; idx += 256)
      Sbf[idx] = (__bf16)Sbuf[idx];
    for (int idx = tid; idx < BT * TDIMC; idx += 256) {
      int row = idx >> 7, j = idx & 127;
      Hbuf[(size_t)row * 640 + 512 + j] = teB[(size_t)t * TDIMC + j];
    }
    __syncthreads();

    // se = gelu(s @ Ws + bs) -> Hbuf cols [0..511]
    gemm_layer<64, 512, 4, true, false>(Sbf, 64, WsT, bs, Hbuf, nullptr, 640);
    __syncthreads();
    // h1 = gelu([se|te] @ Wj1 + bj1)
    gemm_layer<640, 512, 4, true, false>(Hbuf, 640, Wj1T, bj1, H1, nullptr, 512);
    __syncthreads();
    // h2 = gelu(h1 @ Wj2 + bj2)
    gemm_layer<512, 512, 4, true, false>(H1, 512, Wj2T, bj2, H2, nullptr, 512);
    __syncthreads();
    // pfs = h2 @ Wj3 + bj3  (f32 out)
    gemm_layer<512, 128, 1, false, true>(H2, 512, Wj3T, bj3, nullptr, Pf, 128);
    __syncthreads();

    // elementwise tail: 8 threads per row, 8 dims each
    {
      const float t_f  = (float)t;
      const float bstd = sqrtf(DT_F * t_f / (t_f + 1.0f));
      const float bstd_safe = (t > 0) ? bstd : 1.0f;
      float lpf = 0.0f, lpb = 0.0f;
      #pragma unroll
      for (int q = 0; q < 8; ++q) {
        const int d = edg + q;
        float mean  = Pf[erow * 128 + d];
        float rawlv = Pf[erow * 128 + 64 + d];
        float lv    = fast_tanh(rawlv) * LOG_VAR_RANGE_F;   // LOG_PF_VAR = 0
        float sdv   = __expf(0.5f * lv);
        float sc    = Sbuf[erow * 64 + d];
        float snext = sc + DT_F * mean + SQRT_DT_F * sdv * er[q];
        float sw    = wrap_f(snext);
        float dlt   = wrap_f(sw - sc);
        lpf += wrapped_lp(dlt, DT_F * mean, SQRT_DT_F * sdv);
        float bm  = -sw / (t_f + 1.0f);
        float db  = wrap_f(sc - sw);
        float lb  = wrapped_lp(db, bm, bstd_safe);
        lpb += (t > 0) ? lb : 0.0f;
        Sbuf[erow * 64 + d] = sw;
        states[((size_t)(r0 + erow) * (TC + 1) + (t + 1)) * DIMC + d] = sw;
      }
      // width-8 reduction within wave32
      #pragma unroll
      for (int off = 4; off > 0; off >>= 1) {
        lpf += __shfl_xor(lpf, off, 8);
        lpb += __shfl_xor(lpb, off, 8);
      }
      if ((tid & 7) == 0) {
        gLogpf[(size_t)(r0 + erow) * TC + t] = lpf;
        gLogpb[(size_t)(r0 + erow) * TC + t] = lpb;
        gLogf[(size_t)(r0 + erow) * (TC + 1) + t] = flow;
      }
    }
    __syncthreads();
  }
}

// =====================================================================
// host launcher
// =====================================================================
extern "C" void kernel_launch(void* const* d_in, const int* in_sizes, int n_in,
                              void* d_out, int out_size, void* d_ws, size_t ws_size,
                              hipStream_t stream) {
  // inputs (setup_inputs order)
  const float* s    = (const float*)d_in[0];
  const float* eps  = (const float*)d_in[1];
  const float* Wt1  = (const float*)d_in[2];
  const float* bt1  = (const float*)d_in[3];
  const float* Wt2  = (const float*)d_in[4];
  const float* bt2  = (const float*)d_in[5];
  const float* Ws   = (const float*)d_in[6];
  const float* bs   = (const float*)d_in[7];
  const float* Wj1  = (const float*)d_in[8];
  const float* bj1  = (const float*)d_in[9];
  const float* Wj2  = (const float*)d_in[10];
  const float* bj2  = (const float*)d_in[11];
  const float* Wj3  = (const float*)d_in[12];
  const float* bj3  = (const float*)d_in[13];
  const float* flow = (const float*)d_in[14];

  __bf16* ws = (__bf16*)d_ws;
  __bf16* WsT  = ws + WS_WST;
  __bf16* Wj1T = ws + WS_WJ1T;
  __bf16* Wj2T = ws + WS_WJ2T;
  __bf16* Wj3T = ws + WS_WJ3T;
  __bf16* teB  = ws + WS_TE;

  // outputs, concatenated flat
  float* out     = (float*)d_out;
  float* states  = out;
  float* gLogpf  = states + (size_t)BC * (TC + 1) * DIMC;
  float* gLogpb  = gLogpf + (size_t)BC * TC;
  float* gLogf   = gLogpb + (size_t)BC * TC;

  // prep: transpose/convert weights to bf16 [N][K]
  k_transpose_bf16<<<128,  256, 0, stream>>>(Ws,  WsT,  64,  512);
  k_transpose_bf16<<<1280, 256, 0, stream>>>(Wj1, Wj1T, 640, 512);
  k_transpose_bf16<<<1024, 256, 0, stream>>>(Wj2, Wj2T, 512, 512);
  k_transpose_bf16<<<256,  256, 0, stream>>>(Wj3, Wj3T, 512, 128);
  // prep: batch-independent time embedding for all T steps
  k_time_embed<<<TC, 128, 0, stream>>>(Wt1, bt1, Wt2, bt2, teB);

  // main sequential scan (32 blocks x 8 wave32s, 132KB dynamic LDS)
  constexpr size_t SMEM = 135168;
  k_scan<<<NBLK, 256, SMEM, stream>>>(s, eps, WsT, bs, Wj1T, bj1, Wj2T, bj2,
                                      Wj3T, bj3, teB, flow,
                                      states, gLogpf, gLogpb, gLogf);
}
